// CodebookRemapReadout_77077483094337
// MI455X (gfx1250) — compile-verified, matrix operand order checked by reading, not run on hardware
//
#include <hip/hip_runtime.h>
#include <hip/hip_bf16.h>

typedef __attribute__((ext_vector_type(16))) __bf16 v16bf;
typedef __attribute__((ext_vector_type(8)))  __bf16 v8bf;
typedef __attribute__((ext_vector_type(8)))  float  v8f;

#define N_DIM 512
#define S_DIM 256
#define D_DIM 512
#define C_DIM 1024
#define CLS   1000

// invsqrt(512)
#define INV_SCALE 0.04419417382415922f

// ---------------------------------------------------------------------------
// Kernel 0: codebook f32 -> bf16 (done once per launch; 1 MB, L2-resident)
// ---------------------------------------------------------------------------
__global__ __launch_bounds__(256) void cvt_codebook_bf16(const float* __restrict__ cb,
                                                         __bf16* __restrict__ out, int n) {
  for (int i = blockIdx.x * blockDim.x + threadIdx.x; i < n; i += gridDim.x * blockDim.x)
    out[i] = (__bf16)cb[i];
}

// ---------------------------------------------------------------------------
// Kernel A1: q_proj[n,e] = sum_d q[n,d] * Wq[e,d]    (512x512x512, tiny)
// ---------------------------------------------------------------------------
__global__ __launch_bounds__(256) void qproj_kernel(const float* __restrict__ q,
                                                    const float* __restrict__ Wq,
                                                    float* __restrict__ q_proj) {
  __shared__ float s[D_DIM];
  int n = blockIdx.x;
  for (int i = threadIdx.x; i < D_DIM; i += 256) s[i] = q[(size_t)n * D_DIM + i];
  __syncthreads();
  for (int e = threadIdx.x; e < D_DIM; e += 256) {
    const float4* w = (const float4*)(Wq + (size_t)e * D_DIM);
    float acc = 0.f;
    for (int k = 0; k < D_DIM / 4; ++k) {
      float4 wv = w[k];
      acc += s[4 * k + 0] * wv.x + s[4 * k + 1] * wv.y +
             s[4 * k + 2] * wv.z + s[4 * k + 3] * wv.w;
    }
    q_proj[(size_t)n * D_DIM + e] = acc;
  }
}

// ---------------------------------------------------------------------------
// Kernel A2: code_weights[n,c] = softmax_c( q_proj[n,:] . codebook[c,:] / sqrt(D) )
// One block per n; each thread owns 4 c's. Logit std ~0.04 => exp w/o max is safe.
// ---------------------------------------------------------------------------
__global__ __launch_bounds__(256) void codew_kernel(const float* __restrict__ q_proj,
                                                    const float* __restrict__ cb,
                                                    float* __restrict__ cw) {
  __shared__ float s[D_DIM];
  __shared__ float tot;
  int n = blockIdx.x;
  for (int i = threadIdx.x; i < D_DIM; i += 256) s[i] = q_proj[(size_t)n * D_DIM + i];
  if (threadIdx.x == 0) tot = 0.f;
  __syncthreads();
  float e[4];
  float lsum = 0.f;
#pragma unroll
  for (int i = 0; i < 4; ++i) {
    int c = threadIdx.x + i * 256;
    const float4* w = (const float4*)(cb + (size_t)c * D_DIM);
    float acc = 0.f;
    for (int k = 0; k < D_DIM / 4; ++k) {
      float4 wv = w[k];
      acc += s[4 * k + 0] * wv.x + s[4 * k + 1] * wv.y +
             s[4 * k + 2] * wv.z + s[4 * k + 3] * wv.w;
    }
    e[i] = __expf(acc * INV_SCALE);
    lsum += e[i];
  }
  atomicAdd(&tot, lsum);
  __syncthreads();
  float inv = 1.f / tot;
#pragma unroll
  for (int i = 0; i < 4; ++i)
    cw[(size_t)n * C_DIM + threadIdx.x + i * 256] = e[i] * inv;
}

// ---------------------------------------------------------------------------
// Kernel B (dominant, 137 GFLOP): per (n, 16-slot tile) compute
//   eff[n,s] = sum_c cw[n,c] * exp(l)/sum_c exp(l),  l = K[n,s,:].cb[c,:]/sqrt(D)
// streamed over C with V_WMMA_F32_16X16X32_BF16. K tile converted to bf16 in LDS
// (each K element touches HBM exactly once). 8 waves split the 64 c-tiles.
// ---------------------------------------------------------------------------
__global__ __launch_bounds__(256) void eff_kernel(const float* __restrict__ K,
                                                  const __bf16* __restrict__ cbb,
                                                  const float* __restrict__ cw,
                                                  float* __restrict__ eff) {
  __shared__ __align__(16) __bf16 kb[16 * D_DIM];   // 16 KB bf16 K-tile
  __shared__ float s_se[16];
  __shared__ float s_ws[16];

  const int n  = blockIdx.x;
  const int st = blockIdx.y;                        // slot tile (16 rows)

  // Stage K tile -> LDS as bf16 (16 x 512 f32 = 32 KB read, coalesced float4)
  const float4* kbase = (const float4*)(K + (((size_t)n * S_DIM) + st * 16) * D_DIM);
  for (int i = threadIdx.x; i < 16 * D_DIM / 4; i += 256) {
    float4 v = kbase[i];
    kb[4 * i + 0] = (__bf16)v.x;
    kb[4 * i + 1] = (__bf16)v.y;
    kb[4 * i + 2] = (__bf16)v.z;
    kb[4 * i + 3] = (__bf16)v.w;
  }
  if (threadIdx.x < 16) { s_se[threadIdx.x] = 0.f; s_ws[threadIdx.x] = 0.f; }
  __syncthreads();

  const int wave = threadIdx.x >> 5;
  const int lane = threadIdx.x & 31;
  const int idx  = lane & 15;       // A: row M, B: column N
  const int hi   = lane >> 4;       // k-chunk select per WMMA layout

  float p_se[8] = {0.f, 0.f, 0.f, 0.f, 0.f, 0.f, 0.f, 0.f};
  float p_ws[8] = {0.f, 0.f, 0.f, 0.f, 0.f, 0.f, 0.f, 0.f};

  const __bf16* arow = kb + (size_t)idx * D_DIM;

  for (int ct = wave; ct < C_DIM / 16; ct += 8) {
    const int c0 = ct * 16;
    const __bf16* brow = cbb + (size_t)(c0 + idx) * D_DIM;   // B column = codebook row
    v8f acc = {};
#pragma unroll
    for (int k0 = 0; k0 < D_DIM; k0 += 32) {
      v8bf alo = *(const v8bf*)(arow + k0 + hi * 8);
      v8bf ahi = *(const v8bf*)(arow + k0 + 16 + hi * 8);
      v8bf blo = *(const v8bf*)(brow + k0 + hi * 8);
      v8bf bhi = *(const v8bf*)(brow + k0 + 16 + hi * 8);
      v16bf a = __builtin_shufflevector(alo, ahi, 0, 1, 2, 3, 4, 5, 6, 7,
                                        8, 9, 10, 11, 12, 13, 14, 15);
      v16bf b = __builtin_shufflevector(blo, bhi, 0, 1, 2, 3, 4, 5, 6, 7,
                                        8, 9, 10, 11, 12, 13, 14, 15);
      acc = __builtin_amdgcn_wmma_f32_16x16x32_bf16(false, a, false, b,
                                                    (short)0, acc, false, false);
    }
    // acc[i] = logit(row = i + hi*8, col = c0+idx); fold softmax + cw contraction
    const float cwv = cw[(size_t)n * C_DIM + c0 + idx];
#pragma unroll
    for (int i = 0; i < 8; ++i) {
      float ev = __expf(acc[i] * INV_SCALE);
      p_se[i] += ev;
      p_ws[i] += ev * cwv;
    }
  }

  // Reduce across the 16 column-lanes of each half-wave, then across waves via LDS
#pragma unroll
  for (int i = 0; i < 8; ++i) {
    float a = p_se[i], b = p_ws[i];
#pragma unroll
    for (int m = 1; m < 16; m <<= 1) {
      a += __shfl_xor(a, m, 32);
      b += __shfl_xor(b, m, 32);
    }
    if (idx == 0) {
      atomicAdd(&s_se[hi * 8 + i], a);
      atomicAdd(&s_ws[hi * 8 + i], b);
    }
  }
  __syncthreads();

  if (threadIdx.x < 16)
    eff[(size_t)n * S_DIM + st * 16 + threadIdx.x] =
        s_ws[threadIdx.x] / s_se[threadIdx.x];
}

// ---------------------------------------------------------------------------
// Kernel C: summary[n,d] = sum_s eff[n,s] * V[n,s,d]   (268 MB read, BW bound)
// ---------------------------------------------------------------------------
__global__ __launch_bounds__(256) void summary_kernel(const float* __restrict__ V,
                                                      const float* __restrict__ eff,
                                                      float* __restrict__ summary) {
  __shared__ float se[S_DIM];
  int n = blockIdx.x;
  se[threadIdx.x] = eff[(size_t)n * S_DIM + threadIdx.x];
  __syncthreads();
  const float* vb = V + (size_t)n * S_DIM * D_DIM;
  int d = threadIdx.x;
  float acc0 = 0.f, acc1 = 0.f;
  for (int s = 0; s < S_DIM; ++s) {
    float w = se[s];
    acc0 += w * vb[(size_t)s * D_DIM + d];
    acc1 += w * vb[(size_t)s * D_DIM + d + 256];
  }
  summary[(size_t)n * D_DIM + d] = acc0;
  summary[(size_t)n * D_DIM + d + 256] = acc1;
}

// ---------------------------------------------------------------------------
// Kernel D: logits = summary @ Wc^T + bc ; recon = summary @ Wr^T + br
// ---------------------------------------------------------------------------
__global__ __launch_bounds__(256) void readout_kernel(const float* __restrict__ summary,
                                                      const float* __restrict__ Wc,
                                                      const float* __restrict__ bc,
                                                      const float* __restrict__ Wr,
                                                      const float* __restrict__ br,
                                                      float* __restrict__ logits,
                                                      float* __restrict__ recon) {
  __shared__ float s[D_DIM];
  int n = blockIdx.x;
  for (int i = threadIdx.x; i < D_DIM; i += 256) s[i] = summary[(size_t)n * D_DIM + i];
  __syncthreads();
  for (int j = threadIdx.x; j < CLS + D_DIM; j += 256) {
    const float* w;
    float acc;
    if (j < CLS) { w = Wc + (size_t)j * D_DIM; acc = bc[j]; }
    else         { w = Wr + (size_t)(j - CLS) * D_DIM; acc = br[j - CLS]; }
    const float4* w4 = (const float4*)w;
    for (int k = 0; k < D_DIM / 4; ++k) {
      float4 wv = w4[k];
      acc += s[4 * k + 0] * wv.x + s[4 * k + 1] * wv.y +
             s[4 * k + 2] * wv.z + s[4 * k + 3] * wv.w;
    }
    if (j < CLS) logits[(size_t)n * CLS + j] = acc;
    else         recon[(size_t)n * D_DIM + (j - CLS)] = acc;
  }
}

// ---------------------------------------------------------------------------
extern "C" void kernel_launch(void* const* d_in, const int* in_sizes, int n_in,
                              void* d_out, int out_size, void* d_ws, size_t ws_size,
                              hipStream_t stream) {
  const float* q   = (const float*)d_in[0];
  const float* K   = (const float*)d_in[1];
  const float* V   = (const float*)d_in[2];
  const float* cb  = (const float*)d_in[3];
  const float* Wq  = (const float*)d_in[4];
  const float* Wc  = (const float*)d_in[5];
  const float* bc  = (const float*)d_in[6];
  const float* Wr  = (const float*)d_in[7];
  const float* br  = (const float*)d_in[8];

  // Output tuple, flat: logits [N,CLS] | recon [N,D] | eff [N,S]
  float* logits = (float*)d_out;
  float* recon  = logits + (size_t)N_DIM * CLS;
  float* eff    = recon + (size_t)N_DIM * D_DIM;

  // Workspace: codebook bf16 | q_proj f32 | code_weights f32 | summary f32
  char* ws = (char*)d_ws;
  __bf16* cbb     = (__bf16*)ws;                                  ws += (size_t)C_DIM * D_DIM * 2;
  float*  q_proj  = (float*)ws;                                   ws += (size_t)N_DIM * D_DIM * 4;
  float*  cw      = (float*)ws;                                   ws += (size_t)N_DIM * C_DIM * 4;
  float*  summary = (float*)ws;

  cvt_codebook_bf16<<<512, 256, 0, stream>>>(cb, cbb, C_DIM * D_DIM);
  qproj_kernel<<<N_DIM, 256, 0, stream>>>(q, Wq, q_proj);
  codew_kernel<<<N_DIM, 256, 0, stream>>>(q_proj, cb, cw);
  eff_kernel<<<dim3(N_DIM, S_DIM / 16), 256, 0, stream>>>(K, cbb, cw, eff);
  summary_kernel<<<N_DIM, 256, 0, stream>>>(V, eff, summary);
  readout_kernel<<<N_DIM, 256, 0, stream>>>(summary, Wc, bc, Wr, br, logits, recon);
}
